// DeeperGCN_35966056137084
// MI455X (gfx1250) — matrix operand order repaired and checked by reading, not compiled, not run on hardware
//
#include <hip/hip_runtime.h>
#include <hip/hip_bf16.h>
#include <cstdint>

#define N_NODES 50000
#define N_EDGES 625000
#define DIM     128
#define NLAYER  7
#define EPSMSG  1e-7f

typedef __attribute__((ext_vector_type(16))) _Float16 v16h;
typedef __attribute__((ext_vector_type(8)))  _Float16 v8h;
typedef __attribute__((ext_vector_type(8)))  float    v8f;

// ---------------------------------------------------------------- init / encode

__global__ void __launch_bounds__(256) k_init_zero(float* __restrict__ segmax,
                                                   float* __restrict__ num,
                                                   float* __restrict__ den) {
  int i = blockIdx.x * 256 + threadIdx.x;           // i over N*DIM (exact)
  segmax[i] = 0.f; num[i] = 0.f; den[i] = 0.f;
}

__global__ void __launch_bounds__(256) k_atom_encode(const int* __restrict__ x,
                                                     const float* __restrict__ atom_emb,
                                                     float* __restrict__ h) {
  int i = blockIdx.x * 256 + threadIdx.x;           // i over N*DIM (exact)
  int n = i >> 7, d = i & 127;
  float s = 0.f;
#pragma unroll
  for (int f = 0; f < 9; ++f) {
    int v = x[n * 9 + f];
    s += atom_emb[(f * 64 + v) * DIM + d];          // atom_emb[9][64][128]
  }
  h[i] = s;
}

// Swizzle W[l][k][n] (f32) into per-lane-contiguous f16 WMMA B fragments:
// Wp[((l*4+kt)*2+hi)*128 + col][j] = W[l][kt*32 + hi*16 + j][col],  j=0..15
__global__ void __launch_bounds__(256) k_convert_w(const float* __restrict__ W,
                                                   _Float16* __restrict__ Wp) {
  int i = blockIdx.x * 256 + threadIdx.x;           // over 7*4*2*128*16 = 114688
  int j   = i & 15;
  int col = (i >> 4) & 127;
  int hi  = (i >> 11) & 1;
  int kt  = (i >> 12) & 3;
  int l   = i >> 14;
  int k = kt * 32 + hi * 16 + j;
  Wp[i] = (_Float16)W[(l * DIM + k) * DIM + col];
}

// ---------------------------------------------------------------- edge passes
// 256 threads = 8 wave32s per block; one wave per edge; one lane = 4 dims.

__global__ void __launch_bounds__(256) k_edge_max(const int* __restrict__ ei,
                                                  const int* __restrict__ ea,
                                                  const float* __restrict__ bond_emb,
                                                  const float* __restrict__ hin,
                                                  float* __restrict__ segmax) {
  int wave = threadIdx.x >> 5, lane = threadIdx.x & 31;
  int e = blockIdx.x * 8 + wave;                    // exact: E = 78125*8
  int src = ei[e];
  int dst = ei[N_EDGES + e];
  int a0 = ea[e * 3 + 0], a1 = ea[e * 3 + 1], a2 = ea[e * 3 + 2];
  int d0 = lane * 4;
  if (lane == 0) {                                  // stream-ahead prefetch of edge indices
    int pe = min(e + 1024, N_EDGES - 1);
    __builtin_prefetch(ei + pe, 0, 0);
    __builtin_prefetch(ea + (size_t)pe * 3, 0, 0);
  }
  const float4 hv = *(const float4*)(hin + (size_t)src * DIM + d0);
  unsigned* smx = (unsigned*)(segmax + (size_t)dst * DIM + d0);
#pragma unroll
  for (int j = 0; j < 4; ++j) {
    int d = d0 + j;
    float eev = bond_emb[a0 * DIM + d] + bond_emb[1024 + a1 * DIM + d] +
                bond_emb[2048 + a2 * DIM + d];
    float msg = fmaxf(((const float*)&hv)[j] + eev, 0.f) + EPSMSG;
    atomicMax(smx + j, __float_as_uint(msg));       // msg > 0 -> IEEE bits monotonic
  }
}

__global__ void __launch_bounds__(256) k_edge_sum(const int* __restrict__ ei,
                                                  const int* __restrict__ ea,
                                                  const float* __restrict__ bond_emb,
                                                  const float* __restrict__ hin,
                                                  const float* __restrict__ segmax,
                                                  float* __restrict__ num,
                                                  float* __restrict__ den) {
  int wave = threadIdx.x >> 5, lane = threadIdx.x & 31;
  int e = blockIdx.x * 8 + wave;
  int src = ei[e];
  int dst = ei[N_EDGES + e];
  int a0 = ea[e * 3 + 0], a1 = ea[e * 3 + 1], a2 = ea[e * 3 + 2];
  int d0 = lane * 4;
  if (lane == 0) {
    int pe = min(e + 1024, N_EDGES - 1);
    __builtin_prefetch(ei + pe, 0, 0);
  }
  const float4 hv  = *(const float4*)(hin    + (size_t)src * DIM + d0);
  const float4 smv = *(const float4*)(segmax + (size_t)dst * DIM + d0);
  float* np = num + (size_t)dst * DIM + d0;
  float* dp = den + (size_t)dst * DIM + d0;
#pragma unroll
  for (int j = 0; j < 4; ++j) {
    int d = d0 + j;
    float eev = bond_emb[a0 * DIM + d] + bond_emb[1024 + a1 * DIM + d] +
                bond_emb[2048 + a2 * DIM + d];
    float msg = fmaxf(((const float*)&hv)[j] + eev, 0.f) + EPSMSG;
    float ev  = __expf(msg - ((const float*)&smv)[j]);
    atomicAdd(dp + j, ev);
    atomicAdd(np + j, ev * msg);
  }
}

// m = num/den; t = hin + m (f16 A-operand); re-zero accumulators for next layer.
__global__ void __launch_bounds__(256) k_finalize(const float* __restrict__ hin,
                                                  float* __restrict__ num,
                                                  float* __restrict__ den,
                                                  float* __restrict__ segmax,
                                                  _Float16* __restrict__ th) {
  int i = blockIdx.x * 256 + threadIdx.x;           // over N*DIM (exact)
  float m = num[i] / (den[i] + 1e-16f);
  th[i] = (_Float16)(hin[i] + m);
  num[i] = 0.f; den[i] = 0.f; segmax[i] = 0.f;
}

// ---------------------------------------------------------------- WMMA GEMM
// Block = 256 threads = 8 waves; block computes rows [16*bid,16*bid+16), wave w
// owns columns [16w,16w+16). K = 128 in 4 steps of v_wmma_f32_16x16x32_f16.
// A tile staged once per block into LDS via gfx1250 async load-to-LDS
// (ASYNCcnt path), rows padded 16B to spread the fragment reads across banks.

#define APAD 8
#define AROW (DIM + APAD)

#if __has_builtin(__builtin_amdgcn_global_load_async_to_lds_b128)
// Param 1 probed from clang diagnostic: non-const addrspace(1) ptr to int __vector(4).
typedef int b128_t __attribute__((vector_size(16)));
typedef __attribute__((address_space(1))) b128_t* gptr_b128;
typedef __attribute__((address_space(3))) b128_t* lptr_b128;
#endif

template <int RESID>
__global__ void __launch_bounds__(256) k_gemm_wmma(const _Float16* __restrict__ th,
                                                   const _Float16* __restrict__ Wp,
                                                   const float* __restrict__ bias,
                                                   float* __restrict__ h) {
  __shared__ _Float16 As[16 * AROW];                // 16 rows x (128+8) halves
  const int tid  = threadIdx.x;
  const int lane = tid & 31;
  const int wave = tid >> 5;
  const int m    = lane & 15;
  const int hi   = lane >> 4;
  const int col  = wave * 16 + m;
  const size_t row0 = (size_t)blockIdx.x * 16;

  // Cooperative A-tile stage: thread t copies 16B of row (t>>4), segment (t&15).
  {
    const int r = tid >> 4, seg = tid & 15;
    const _Float16* gsrc = th + (row0 + r) * DIM + seg * 8;
    _Float16* ldst = As + r * AROW + seg * 8;
#if __has_builtin(__builtin_amdgcn_global_load_async_to_lds_b128)
    __builtin_amdgcn_global_load_async_to_lds_b128(
        (gptr_b128)(uintptr_t)gsrc, (lptr_b128)(uintptr_t)ldst, 0, 0);
    asm volatile("s_wait_asynccnt 0x0" ::: "memory");
#else
    *(v8h*)ldst = *(const v8h*)gsrc;                // fallback: reg-staged LDS copy
#endif
  }
  __syncthreads();

  const _Float16* arow = As + m * AROW;
  v8f acc = {};
#pragma unroll
  for (int kt = 0; kt < 4; ++kt) {
    // A 16x32 f16 fragment: lanes 0-15 hold K {0..15}, lanes 16-31 {16..31}
    const int kbase = kt * 32 + hi * 16;
    v8h alo = *(const v8h*)(arow + kbase);          // ds_load_b128
    v8h ahi = *(const v8h*)(arow + kbase + 8);
    v16h a;
#pragma unroll
    for (int i = 0; i < 8; ++i) { a[i] = alo[i]; a[8 + i] = ahi[i]; }
    // B 32x16 fragment: pre-swizzled, one contiguous 32B load per lane
    v16h bfrag = *(const v16h*)(Wp + ((((kt * 2 + hi) * 128) + col) << 4));
    acc = __builtin_amdgcn_wmma_f32_16x16x32_f16(false, a, false, bfrag,
                                                 (short)0, acc, false, false);
  }

  const float bc = bias[col];
  float res[8];
  if (RESID) {                                      // compile-time: unconditional loads
#pragma unroll
    for (int r = 0; r < 8; ++r)
      res[r] = h[(row0 + hi * 8 + r) * DIM + col];
  }
#pragma unroll
  for (int r = 0; r < 8; ++r) {                     // C/D: VGPR r -> M = r + 8*hi
    size_t o = (row0 + hi * 8 + r) * DIM + col;
    float v = acc[r] + bc;
    if (RESID) v += res[r];
    h[o] = v;
  }
}

// ---------------------------------------------------------------- LayerNorm (+ReLU)
// One wave per node, lane = 4 dims, wave32 shuffle reduction.

__global__ void __launch_bounds__(256) k_ln(const float* __restrict__ hin,
                                            const float* __restrict__ gamma,
                                            const float* __restrict__ beta,
                                            float* __restrict__ hout,
                                            int do_relu) {
  int wave = threadIdx.x >> 5, lane = threadIdx.x & 31;
  size_t n = (size_t)blockIdx.x * 8 + wave;         // exact: N = 6250*8
  int d0 = lane * 4;
  float4 v = *(const float4*)(hin + n * DIM + d0);
  float s  = v.x + v.y + v.z + v.w;
  float s2 = v.x * v.x + v.y * v.y + v.z * v.z + v.w * v.w;
#pragma unroll
  for (int off = 16; off; off >>= 1) {
    s  += __shfl_xor(s,  off, 32);
    s2 += __shfl_xor(s2, off, 32);
  }
  float mu  = s * (1.f / 128.f);
  float var = s2 * (1.f / 128.f) - mu * mu;
  float rs  = rsqrtf(var + 1e-5f);
  float4 g  = *(const float4*)(gamma + d0);
  float4 bt = *(const float4*)(beta + d0);
  float4 o;
  o.x = (v.x - mu) * rs * g.x + bt.x;
  o.y = (v.y - mu) * rs * g.y + bt.y;
  o.z = (v.z - mu) * rs * g.z + bt.z;
  o.w = (v.w - mu) * rs * g.w + bt.w;
  if (do_relu) {
    o.x = fmaxf(o.x, 0.f); o.y = fmaxf(o.y, 0.f);
    o.z = fmaxf(o.z, 0.f); o.w = fmaxf(o.w, 0.f);
  }
  *(float4*)(hout + n * DIM + d0) = o;
}

// ---------------------------------------------------------------- driver

extern "C" void kernel_launch(void* const* d_in, const int* in_sizes, int n_in,
                              void* d_out, int out_size, void* d_ws, size_t ws_size,
                              hipStream_t stream) {
  (void)in_sizes; (void)n_in; (void)out_size; (void)ws_size;
  const int*   x        = (const int*)d_in[0];     // [N,9]
  const int*   ei       = (const int*)d_in[1];     // [2,E]
  const int*   ea       = (const int*)d_in[2];     // [E,3]
  const float* atom_emb = (const float*)d_in[3];   // [9,64,128]
  const float* bond_emb = (const float*)d_in[4];   // [3,8,128]
  const float* W        = (const float*)d_in[5];   // [7,128,128]
  const float* b        = (const float*)d_in[6];   // [7,128]
  const float* gamma    = (const float*)d_in[7];   // [7,128]
  const float* beta     = (const float*)d_in[8];   // [7,128]
  float*       out      = (float*)d_out;           // [N,128] f32

  char* ws = (char*)d_ws;
  size_t off = 0;
  auto alloc = [&](size_t bytes) -> char* {
    char* p = ws + off;
    off = (off + bytes + 255) & ~(size_t)255;
    return p;
  };
  const size_t ND = (size_t)N_NODES * DIM;
  float*    h      = (float*)alloc(ND * 4);
  float*    h2     = (float*)alloc(ND * 4);
  float*    segmax = (float*)alloc(ND * 4);
  float*    num    = (float*)alloc(ND * 4);
  float*    den    = (float*)alloc(ND * 4);
  _Float16* th     = (_Float16*)alloc(ND * 2);
  _Float16* Wp     = (_Float16*)alloc((size_t)NLAYER * 16384 * 2);

  dim3 blk(256);
  k_init_zero<<<ND / 256, blk, 0, stream>>>(segmax, num, den);
  k_convert_w<<<(NLAYER * 16384) / 256, blk, 0, stream>>>(W, Wp);
  k_atom_encode<<<ND / 256, blk, 0, stream>>>(x, atom_emb, h);

  for (int l = 0; l < NLAYER; ++l) {
    const float* hin = (l == 0) ? h : h2;
    k_edge_max<<<N_EDGES / 8, blk, 0, stream>>>(ei, ea, bond_emb, hin, segmax);
    k_edge_sum<<<N_EDGES / 8, blk, 0, stream>>>(ei, ea, bond_emb, hin, segmax, num, den);
    k_finalize<<<ND / 256, blk, 0, stream>>>(hin, num, den, segmax, th);
    if (l == 0)
      k_gemm_wmma<0><<<N_NODES / 16, blk, 0, stream>>>(th, Wp + (size_t)l * 16384,
                                                       b + l * DIM, h);
    else
      k_gemm_wmma<1><<<N_NODES / 16, blk, 0, stream>>>(th, Wp + (size_t)l * 16384,
                                                       b + l * DIM, h);
    if (l < NLAYER - 1)
      k_ln<<<N_NODES / 8, blk, 0, stream>>>(h, gamma + l * DIM, beta + l * DIM, h2, 1);
  }
  k_ln<<<N_NODES / 8, blk, 0, stream>>>(h, gamma + (NLAYER - 1) * DIM,
                                        beta + (NLAYER - 1) * DIM, out, 0);
}
